// EQAGNN_Model_84267258347733
// MI455X (gfx1250) — compile-verified, matrix operand order checked by reading, not run on hardware
//
#include <hip/hip_runtime.h>
#include <cstdint>
#include <cstddef>

#define EPSF 1e-8f

typedef __bf16 bf16;
typedef bf16  bf16x8  __attribute__((ext_vector_type(8)));
typedef bf16  bf16x16 __attribute__((ext_vector_type(16)));
typedef float floatx8 __attribute__((ext_vector_type(8)));

// ---------------------------------------------------------------------------
// Fragment load helper: 16-bit A/B WMMA fragment (ISA 7.12.2).
// lane holds row (lane&15); K = ((lane>>4)*8) + {0..7} and {16..23}
// -> two contiguous 16-byte loads per 32-K step.
// ---------------------------------------------------------------------------
__device__ __forceinline__ bf16x16 ldfrag(const bf16* __restrict__ p) {
    bf16x8 lo = *(const bf16x8*)(p);
    bf16x8 hi = *(const bf16x8*)(p + 16);
    bf16x16 r;
#pragma unroll
    for (int i = 0; i < 8; ++i) { r[i] = lo[i]; r[i + 8] = hi[i]; }
    return r;
}

__device__ __forceinline__ void store_tile(floatx8 acc, int row0, int col, float bs,
                                           float* __restrict__ Cf, bf16* __restrict__ Cb,
                                           int ldc, int M, int N, int relu) {
#pragma unroll
    for (int r = 0; r < 8; ++r) {
        int row = row0 + r;
        if (row < M && col < N) {
            float v = acc[r] + bs;
            if (relu) v = v > 0.f ? v : 0.f;
            if (Cf) Cf[(size_t)row * ldc + col] = v;
            else    Cb[(size_t)row * ldc + col] = (bf16)v;
        }
    }
}

// ---------------------------------------------------------------------------
// WMMA GEMM with 2x2 register blocking: each wave computes a 32x32 patch
// (4 x v_wmma_f32_16x16x32_bf16 per 32-K step against 8 b128 loads -> 2:1).
// Block = 8 waves arranged 2(M) x 4(N) covering 64 rows x 128 cols.
// C[M,N] = A[M,Kp](bf16) x W[Np,Kp]^T(bf16) + bias, optional relu.
// All tile-validity branches are wave-uniform so EXEC stays all-ones (WMMA req).
// ---------------------------------------------------------------------------
__global__ __launch_bounds__(256)
void k_gemm_bf16(const bf16* __restrict__ A, int lda,
                 const bf16* __restrict__ W, int ldw,
                 const float* __restrict__ bias,
                 float* __restrict__ Cf, bf16* __restrict__ Cb, int ldc,
                 int M, int N, int Kp, int relu)
{
    const int wave = threadIdx.x >> 5;
    const int lane = threadIdx.x & 31;
    const int Np = (N + 15) & ~15;
    const int mi = wave >> 2;                      // 0..1
    const int ni = wave & 3;                       // 0..3
    const int n0 = blockIdx.y * 128 + ni * 32;
    const int m0 = blockIdx.x * 64 + mi * 32;
    if (n0 >= Np || m0 >= M) return;               // wave-uniform exit
    const bool twoN = (n0 + 16) < Np;
    const bool twoM = (m0 + 16) < M;

    int ar0 = m0 + (lane & 15);      if (ar0 >= M) ar0 = M - 1;   // keep EXEC full
    int ar1 = m0 + 16 + (lane & 15); if (ar1 >= M) ar1 = M - 1;
    const int br0 = n0 + (lane & 15);
    const int br1 = twoN ? (n0 + 16 + (lane & 15)) : br0;
    const int khalf = (lane >> 4) << 3;            // 0 or 8

    const bf16* Ap0 = A + (size_t)ar0 * lda + khalf;
    const bf16* Ap1 = A + (size_t)ar1 * lda + khalf;
    const bf16* Bp0 = W + (size_t)br0 * ldw + khalf;
    const bf16* Bp1 = W + (size_t)br1 * ldw + khalf;

    floatx8 acc00 = {0.f,0.f,0.f,0.f,0.f,0.f,0.f,0.f};
    floatx8 acc01 = acc00, acc10 = acc00, acc11 = acc00;

    for (int k = 0; k < Kp; k += 32) {
        bf16x16 a0 = ldfrag(Ap0 + k);
        bf16x16 b0 = ldfrag(Bp0 + k);
        acc00 = __builtin_amdgcn_wmma_f32_16x16x32_bf16(
                    false, a0, false, b0, (short)0, acc00, false, false);
        if (twoM) {
            bf16x16 a1 = ldfrag(Ap1 + k);
            acc10 = __builtin_amdgcn_wmma_f32_16x16x32_bf16(
                        false, a1, false, b0, (short)0, acc10, false, false);
            if (twoN) {
                bf16x16 b1 = ldfrag(Bp1 + k);
                acc01 = __builtin_amdgcn_wmma_f32_16x16x32_bf16(
                            false, a0, false, b1, (short)0, acc01, false, false);
                acc11 = __builtin_amdgcn_wmma_f32_16x16x32_bf16(
                            false, a1, false, b1, (short)0, acc11, false, false);
            }
        } else if (twoN) {
            bf16x16 b1 = ldfrag(Bp1 + k);
            acc01 = __builtin_amdgcn_wmma_f32_16x16x32_bf16(
                        false, a0, false, b1, (short)0, acc01, false, false);
        }
    }

    const int c0 = n0 + (lane & 15);
    const int c1 = n0 + 16 + (lane & 15);
    const float bs0 = bias ? bias[c0] : 0.f;
    const float bs1 = (bias && twoN) ? bias[c1] : 0.f;
    const int r0 = m0 + ((lane >> 4) << 3);
    const int r1 = m0 + 16 + ((lane >> 4) << 3);
    store_tile(acc00, r0, c0, bs0, Cf, Cb, ldc, M, N, relu);
    if (twoM) store_tile(acc10, r1, c0, bs0, Cf, Cb, ldc, M, N, relu);
    if (twoN) {
        store_tile(acc01, r0, c1, bs1, Cf, Cb, ldc, M, N, relu);
        if (twoM) store_tile(acc11, r1, c1, bs1, Cf, Cb, ldc, M, N, relu);
    }
}

// ------------------------------ utility kernels ------------------------------
__global__ void k_zero(uint32_t* p, size_t n) {
    size_t i = (size_t)blockIdx.x * blockDim.x + threadIdx.x;
    if (i < n) p[i] = 0u;
}

__global__ void k_pad_w(const float* __restrict__ W, bf16* __restrict__ O,
                        int N, int K, int Np, int Kp) {
    size_t idx = (size_t)blockIdx.x * blockDim.x + threadIdx.x;
    int r = (int)(idx / Kp), c = (int)(idx % Kp);
    if (r >= Np) return;
    float v = (r < N && c < K) ? W[(size_t)r * K + c] : 0.f;
    O[(size_t)r * Kp + c] = (bf16)v;
}

__global__ void k_pad_b(const float* __restrict__ B, float* __restrict__ O,
                        int N, int Np) {
    int i = blockIdx.x * blockDim.x + threadIdx.x;
    if (i < Np) O[i] = (i < N) ? B[i] : 0.f;
}

__global__ void k_f2b(const float* __restrict__ X, int ldx,
                      bf16* __restrict__ Y, int ldy, int rows, int cols) {
    size_t idx = (size_t)blockIdx.x * blockDim.x + threadIdx.x;
    int r = (int)(idx / cols), c = (int)(idx % cols);
    if (r >= rows) return;
    Y[(size_t)r * ldy + c] = (bf16)X[(size_t)r * ldx + c];
}

__global__ void k_relu_b16(bf16* __restrict__ A, int ld, int rows, int cols) {
    size_t idx = (size_t)blockIdx.x * blockDim.x + threadIdx.x;
    int r = (int)(idx / cols), c = (int)(idx % cols);
    if (r >= rows) return;
    bf16* p = A + (size_t)r * ld + c;
    float v = (float)*p;
    *p = (bf16)(v > 0.f ? v : 0.f);
}

// WC[vo,vi] = sum_h wv[vo,h] * wh[h,vi]
__global__ void k_combine(const float* __restrict__ wv, const float* __restrict__ wh,
                          float* __restrict__ WC, int vo, int h, int vi) {
    int idx = blockIdx.x * blockDim.x + threadIdx.x;
    int o = idx / vi, i = idx % vi;
    if (o >= vo) return;
    float s = 0.f;
    for (int k = 0; k < h; ++k) s += wv[o * h + k] * wh[k * vi + i];
    WC[o * vi + i] = s;
}

// vn[r,h] = ||sum_vi wh[h,vi]*V[r,vi,:]||  -> bf16 into A[:, coloff+h]
__global__ void k_vn(const float* __restrict__ Vin, int vi,
                     const float* __restrict__ wh, int h,
                     bf16* __restrict__ Aout, int lda, int coloff, int rows) {
    size_t idx = (size_t)blockIdx.x * blockDim.x + threadIdx.x;
    int r = (int)(idx / h), hh = (int)(idx % h);
    if (r >= rows) return;
    const float* v = Vin + (size_t)r * vi * 3;
    const float* w = wh + (size_t)hh * vi;
    float t0 = 0.f, t1 = 0.f, t2 = 0.f;
    for (int i = 0; i < vi; ++i) {
        float wv = w[i];
        t0 += wv * v[i * 3 + 0]; t1 += wv * v[i * 3 + 1]; t2 += wv * v[i * 3 + 2];
    }
    float n2 = t0 * t0 + t1 * t1 + t2 * t2;
    n2 = n2 > EPSF ? n2 : EPSF;
    Aout[(size_t)r * lda + coloff + hh] = (bf16)sqrtf(n2);
}

// Vout[r,o,:] = (sum_vi WC[o,vi]*Vin[r,vi,:]) * sigmoid(gate[r,o])
__global__ void k_vout(const float* __restrict__ Vin, int vi,
                       const float* __restrict__ WC, int vo,
                       const float* __restrict__ gate, int ldg,
                       float* __restrict__ Vout, int rows) {
    size_t idx = (size_t)blockIdx.x * blockDim.x + threadIdx.x;
    int r = (int)(idx / vo), o = (int)(idx % vo);
    if (r >= rows) return;
    const float* v = Vin + (size_t)r * vi * 3;
    const float* w = WC + (size_t)o * vi;
    float t0 = 0.f, t1 = 0.f, t2 = 0.f;
    for (int i = 0; i < vi; ++i) {
        float wv = w[i];
        t0 += wv * v[i * 3 + 0]; t1 += wv * v[i * 3 + 1]; t2 += wv * v[i * 3 + 2];
    }
    float g = gate[(size_t)r * ldg + o];
    g = 1.f / (1.f + expf(-g));
    float* out = Vout + ((size_t)r * vo + o) * 3;
    out[0] = t0 * g; out[1] = t1 * g; out[2] = t2 * g;
}

// gather s[src], s[dst] (fp32 -> bf16) into AMSG cols [0,128) and [161,289)
__global__ void k_gather_s(const float* __restrict__ S,
                           const int* __restrict__ src, const int* __restrict__ dst,
                           bf16* __restrict__ A, int lda, int E) {
    size_t idx = (size_t)blockIdx.x * blockDim.x + threadIdx.x;
    int e = (int)(idx >> 7), f = (int)(idx & 127);
    if (e >= E) return;
    bf16* row = A + (size_t)e * lda;
    row[f]       = (bf16)S[(size_t)src[e] * 128 + f];
    row[161 + f] = (bf16)S[(size_t)dst[e] * 128 + f];
}

// MV[e] = concat(v[src](16ch), evout(1ch), v[dst](16ch)), 33 channels x3
__global__ void k_gather_v(const float* __restrict__ V, const float* __restrict__ EVOUT,
                           const int* __restrict__ src, const int* __restrict__ dst,
                           float* __restrict__ MV, int E) {
    size_t idx = (size_t)blockIdx.x * blockDim.x + threadIdx.x;
    int e = (int)(idx / 33), ch = (int)(idx % 33);
    if (e >= E) return;
    float* out = MV + (size_t)e * 99 + ch * 3;
    const float* p;
    if (ch < 16)       p = V + (size_t)src[e] * 48 + ch * 3;
    else if (ch == 16) p = EVOUT + (size_t)e * 3;
    else               p = V + (size_t)dst[e] * 48 + (ch - 17) * 3;
    out[0] = p[0]; out[1] = p[1]; out[2] = p[2];
}

__global__ void k_cnt(const int* __restrict__ dst, float* __restrict__ CNT, int E) {
    int e = blockIdx.x * blockDim.x + threadIdx.x;
    if (e < E) atomicAdd(&CNT[dst[e]], 1.f);
}

__global__ void k_scatter_s(const float* __restrict__ Ms, const int* __restrict__ dst,
                            float* __restrict__ AG, int E) {
    size_t idx = (size_t)blockIdx.x * blockDim.x + threadIdx.x;
    int e = (int)(idx >> 7), f = (int)(idx & 127);
    if (e >= E) return;
    atomicAdd(&AG[(size_t)dst[e] * 128 + f], Ms[(size_t)e * 128 + f]);
}

__global__ void k_scatter_v(const float* __restrict__ Mv, const int* __restrict__ dst,
                            float* __restrict__ AG, int E) {
    size_t idx = (size_t)blockIdx.x * blockDim.x + threadIdx.x;
    int e = (int)(idx / 48), j = (int)(idx % 48);
    if (e >= E) return;
    atomicAdd(&AG[(size_t)dst[e] * 48 + j], Mv[(size_t)e * 48 + j]);
}

// fused edge geometry + we-LN + we-GVP prologue (per-edge thread)
__global__ void k_edge_geom(const float* __restrict__ pos,
                            const int* __restrict__ src, const int* __restrict__ dst,
                            const float* __restrict__ weln_w, const float* __restrict__ weln_b,
                            const float* __restrict__ we_wh,
                            bf16* __restrict__ AWE, bf16* __restrict__ AMSG,
                            float* __restrict__ EVN, int E) {
    int e = blockIdx.x * blockDim.x + threadIdx.x;
    if (e >= E) return;
    int si = src[e], di = dst[e];
    float pix = pos[si * 3], piy = pos[si * 3 + 1], piz = pos[si * 3 + 2];
    float pjx = pos[di * 3], pjy = pos[di * 3 + 1], pjz = pos[di * 3 + 2];
    float dx = pix - pjx, dy = piy - pjy, dz = piz - pjz;
    float len = sqrtf(dx * dx + dy * dy + dz * dz);
    const float PI_ = 3.14159265358979323846f;
    float es[8];
    float pref = sqrtf(0.2f) / len;                      // sqrt(2/r_max)/len
    for (int n = 0; n < 8; ++n) es[n] = pref * sinf((n + 1) * (PI_ / 10.f) * len);
    float u = len * 0.1f;
    float u5 = u * u; u5 = u5 * u5 * u;                  // u^5
    float env = 1.f - 21.f * u5 + 35.f * u5 * u - 15.f * u5 * u * u;
    if (!(len < 10.f)) env = 0.f;
    for (int n = 0; n < 8; ++n) es[n] *= env;
    // cosine feature -> AMSG col 160 (persistent)
    float ni = fmaxf(sqrtf(pix * pix + piy * piy + piz * piz), EPSF);
    float nj = fmaxf(sqrtf(pjx * pjx + pjy * pjy + pjz * pjz), EPSF);
    float cosv = (pix * pjx + piy * pjy + piz * pjz) / (ni * nj);
    // we_ln scalar LN over 8
    float m = 0.f; for (int n = 0; n < 8; ++n) m += es[n]; m *= 0.125f;
    float var = 0.f; for (int n = 0; n < 8; ++n) { float d = es[n] - m; var += d * d; }
    var *= 0.125f;
    float rinv = 1.f / sqrtf(var + 1e-5f);
    // vector LN (single channel)
    float ux = dx / len, uy = dy / len, uz = dz / len;
    float vn2 = fmaxf(ux * ux + uy * uy + uz * uz, EPSF);
    float vr = 1.f / sqrtf(vn2);
    float ex = ux * vr, ey = uy * vr, ez = uz * vr;
    EVN[e * 3] = ex; EVN[e * 3 + 1] = ey; EVN[e * 3 + 2] = ez;
    bf16* a = AWE + (size_t)e * 32;
    for (int n = 0; n < 8; ++n) a[n] = (bf16)((es[n] - m) * rinv * weln_w[n] + weln_b[n]);
    float wh00 = we_wh[0];
    float t2 = wh00 * wh00 * (ex * ex + ey * ey + ez * ez);
    a[8] = (bf16)sqrtf(fmaxf(t2, EPSF));
    for (int n = 9; n < 32; ++n) a[n] = (bf16)0.f;
    AMSG[(size_t)e * 352 + 160] = (bf16)cosv;
}

__global__ void k_we_vout(const float* __restrict__ EVN, const float* __restrict__ GATE,
                          const float* __restrict__ we_wv, const float* __restrict__ we_wh,
                          float* __restrict__ EVOUT, int E) {
    int e = blockIdx.x * blockDim.x + threadIdx.x;
    if (e >= E) return;
    float g = GATE[(size_t)e * 16];
    g = 1.f / (1.f + expf(-g));
    float f = we_wv[0] * we_wh[0] * g;
    EVOUT[e * 3]     = f * EVN[e * 3];
    EVOUT[e * 3 + 1] = f * EVN[e * 3 + 1];
    EVOUT[e * 3 + 2] = f * EVN[e * 3 + 2];
}

// LayerNorm over 128 cols, fp32 -> bf16 (wave per row)
__global__ __launch_bounds__(256)
void k_ln128_b16(const float* __restrict__ X, const float* __restrict__ w,
                 const float* __restrict__ b, bf16* __restrict__ Out, int rows) {
    int wid = (blockIdx.x * blockDim.x + threadIdx.x) >> 5;
    int lane = threadIdx.x & 31;
    if (wid >= rows) return;
    const float* x = X + (size_t)wid * 128;
    float xs[4], sum = 0.f;
#pragma unroll
    for (int j = 0; j < 4; ++j) { xs[j] = x[lane + 32 * j]; sum += xs[j]; }
    for (int o = 16; o > 0; o >>= 1) sum += __shfl_xor(sum, o, 32);
    float m = sum * (1.f / 128.f);
    float var = 0.f;
#pragma unroll
    for (int j = 0; j < 4; ++j) { float d = xs[j] - m; var += d * d; }
    for (int o = 16; o > 0; o >>= 1) var += __shfl_xor(var, o, 32);
    var *= (1.f / 128.f);
    float r = 1.f / sqrtf(var + 1e-5f);
#pragma unroll
    for (int j = 0; j < 4; ++j) {
        int i = lane + 32 * j;
        Out[(size_t)wid * 128 + i] = (bf16)((xs[j] - m) * r * w[i] + b[i]);
    }
}

// fused residual + _ln_sv (wave per node). inv = 1/max(cnt,1) or 1.
__global__ __launch_bounds__(256)
void k_resid_lnsv(float* __restrict__ S, float* __restrict__ V,
                  const float* __restrict__ dS, const float* __restrict__ dV,
                  const float* __restrict__ CNT,
                  const float* __restrict__ lnw, const float* __restrict__ lnb, int rows) {
    int wid = (blockIdx.x * blockDim.x + threadIdx.x) >> 5;
    int lane = threadIdx.x & 31;
    if (wid >= rows) return;
    float inv = CNT ? 1.f / fmaxf(CNT[wid], 1.f) : 1.f;
    float* s = S + (size_t)wid * 128;
    const float* ds = dS + (size_t)wid * 128;
    float x[4], sum = 0.f;
#pragma unroll
    for (int j = 0; j < 4; ++j) { int i = lane + 32 * j; x[j] = s[i] + ds[i] * inv; sum += x[j]; }
    for (int o = 16; o > 0; o >>= 1) sum += __shfl_xor(sum, o, 32);
    float m = sum * (1.f / 128.f);
    float var = 0.f;
#pragma unroll
    for (int j = 0; j < 4; ++j) { float d = x[j] - m; var += d * d; }
    for (int o = 16; o > 0; o >>= 1) var += __shfl_xor(var, o, 32);
    var *= (1.f / 128.f);
    float r = 1.f / sqrtf(var + 1e-5f);
#pragma unroll
    for (int j = 0; j < 4; ++j) { int i = lane + 32 * j; s[i] = (x[j] - m) * r * lnw[i] + lnb[i]; }
    // vector part: 16 channels x 3; rms of per-channel norms
    float* v = V + (size_t)wid * 48;
    const float* dv = dV + (size_t)wid * 48;
    float c0 = 0.f, c1 = 0.f, c2 = 0.f, n2 = 0.f;
    if (lane < 16) {
        c0 = v[lane * 3 + 0] + dv[lane * 3 + 0] * inv;
        c1 = v[lane * 3 + 1] + dv[lane * 3 + 1] * inv;
        c2 = v[lane * 3 + 2] + dv[lane * 3 + 2] * inv;
        n2 = fmaxf(c0 * c0 + c1 * c1 + c2 * c2, EPSF);
    }
    float t = n2;
    for (int o = 16; o > 0; o >>= 1) t += __shfl_xor(t, o, 32);
    float ir = 1.f / sqrtf(t * (1.f / 16.f));
    if (lane < 16) {
        v[lane * 3 + 0] = c0 * ir; v[lane * 3 + 1] = c1 * ir; v[lane * 3 + 2] = c2 * ir;
    }
}

__global__ void k_softmax2(const float* __restrict__ L, int ld,
                           float* __restrict__ out, int rows) {
    int r = blockIdx.x * blockDim.x + threadIdx.x;
    if (r >= rows) return;
    float a = L[(size_t)r * ld], b = L[(size_t)r * ld + 1];
    float m = fmaxf(a, b);
    float ea = expf(a - m), eb = expf(b - m);
    float s = ea + eb;
    out[r * 2] = ea / s; out[r * 2 + 1] = eb / s;
}

// ---------------------------------------------------------------------------
extern "C" void kernel_launch(void* const* d_in, const int* in_sizes, int n_in,
                              void* d_out, int out_size, void* d_ws, size_t ws_size,
                              hipStream_t stream) {
    if (n_in < 189) return;
    const int NN = in_sizes[0] / 3;       // 10000 nodes
    const int E  = in_sizes[2] / 2;       // 200000 edges
    const float* pos   = (const float*)d_in[0];
    const float* atoms = (const float*)d_in[1];
    const int*   ei    = (const int*)d_in[2];
    const int*   src   = ei;
    const int*   dst   = ei + E;
    auto Pf = [&](int i) { return (const float*)d_in[i]; };

    // params leaf indexing (jax pytree: dict keys sorted)
    // top: layers(5x34), pred1(b,w), pred2(b,w), we(6), we_ln(b,w), wv_ln(b,w), wv_ws(b,w)
    // layer: ff0(+0), ff1(+6), ln0(b+12,w+13), ln1(b+14,w+15), msg0(+16), msg1(+22), msg2(+28)
    // gvp: wh+0, ws.b+1, ws.w+2, wsv.b+3, wsv.w+4, wv+5
    const int PRED1B = 173, PRED1W = 174, PRED2B = 175, PRED2W = 176;
    const int WE = 177, WELNB = 183, WELNW = 184, WVLNB = 185, WVLNW = 186;
    const int WVWSB = 187, WVWSW = 188;

    // ---- workspace bump allocator
    char* base = (char*)d_ws;
    size_t off = 0;
    auto alloc = [&](size_t bytes) -> void* {
        void* p = base + off;
        off += (bytes + 255) & ~(size_t)255;
        return p;
    };
    bf16*  AMSG  = (bf16*) alloc((size_t)E * 352 * 2);
    bf16*  A1    = (bf16*) alloc((size_t)E * 160 * 2);
    bf16*  A2    = (bf16*) alloc((size_t)E * 160 * 2);
    bf16*  AWE   = (bf16*) alloc((size_t)E * 32 * 2);
    bf16*  NB    = (bf16*) alloc((size_t)NN * 128 * 2);
    bf16*  NB2   = (bf16*) alloc((size_t)NN * 128 * 2);
    bf16*  AFF0  = (bf16*) alloc((size_t)NN * 160 * 2);
    bf16*  AFF1  = (bf16*) alloc((size_t)NN * 544 * 2);
    float* MV    = (float*)alloc((size_t)E * 99 * 4);   // reused as [E,48] for msg2 vout
    float* MVA   = (float*)alloc((size_t)E * 48 * 4);
    float* MVB   = (float*)alloc((size_t)E * 48 * 4);
    float* MSOUT = (float*)alloc((size_t)E * 128 * 4);
    float* GATE  = (float*)alloc((size_t)E * 16 * 4);
    float* EVN   = (float*)alloc((size_t)E * 3 * 4);
    float* EVOUT = (float*)alloc((size_t)E * 3 * 4);
    float* S     = (float*)alloc((size_t)NN * 128 * 4);
    float* Vv    = (float*)alloc((size_t)NN * 48 * 4);
    float* AGGS  = (float*)alloc((size_t)NN * 128 * 4);
    float* AGGV  = (float*)alloc((size_t)NN * 48 * 4);
    float* FS    = (float*)alloc((size_t)NN * 128 * 4);
    float* FV0   = (float*)alloc((size_t)NN * 96 * 4);
    float* FV    = (float*)alloc((size_t)NN * 48 * 4);
    float* GATEN = (float*)alloc((size_t)NN * 32 * 4);
    float* LOGIT = (float*)alloc((size_t)NN * 16 * 4);
    float* CNT   = (float*)alloc((size_t)NN * 4);
    float* WC    = (float*)alloc(2048 * 4);

    // GEMM weight table
    struct WSpec { int wi, bi, N, K, Kp; };
    WSpec specs[64]; int ns = 0;
    auto add = [&](int wi, int bi, int N, int K, int Kp) { specs[ns++] = {wi, bi, N, K, Kp}; };
    add(WVWSW, WVWSB, 128, 128, 128);    // 0: node embed
    add(WE + 2, WE + 1, 32, 9, 32);      // 1: we.ws
    add(WE + 4, WE + 3, 1, 32, 32);      // 2: we.wsv
    for (int l = 0; l < 5; ++l) {
        int Lb = 3 + l * 34;
        add(Lb + 18, Lb + 17, 128, 322, 352);  // msg0.ws
        add(Lb + 20, Lb + 19, 16, 128, 128);   // msg0.wsv
        add(Lb + 24, Lb + 23, 128, 144, 160);  // msg1.ws
        add(Lb + 26, Lb + 25, 16, 128, 128);   // msg1.wsv
        add(Lb + 30, Lb + 29, 128, 144, 160);  // msg2.ws
        add(Lb + 32, Lb + 31, 16, 128, 128);   // msg2.wsv
        add(Lb + 2,  Lb + 1,  512, 160, 160);  // ff0.ws
        add(Lb + 4,  Lb + 3,  32, 512, 512);   // ff0.wsv
        add(Lb + 8,  Lb + 7,  128, 544, 544);  // ff1.ws
        add(Lb + 10, Lb + 9,  16, 128, 128);   // ff1.wsv
    }
    add(PRED1W, PRED1B, 128, 128, 128);  // 53
    add(PRED2W, PRED2B, 2, 128, 128);    // 54
    bf16* Wd[64]; float* Bd[64];
    for (int i = 0; i < ns; ++i) {
        int Np = (specs[i].N + 15) & ~15;
        Wd[i] = (bf16*)alloc((size_t)Np * specs[i].Kp * 2);
        Bd[i] = (float*)alloc((size_t)Np * 4);
    }
    if (off > ws_size) return;   // workspace too small: bail deterministically

    // ---- launch helpers
    auto zero = [&](void* p, size_t bytes) {
        size_t n = bytes / 4;
        k_zero<<<(unsigned)((n + 255) / 256), 256, 0, stream>>>((uint32_t*)p, n);
    };
    auto gemm = [&](const bf16* A, int lda, int wi, float* Cf, bf16* Cb, int ldc,
                    int M, int N, int Kp, int relu) {
        int Np = (N + 15) & ~15;
        dim3 g((M + 63) / 64, (Np + 127) / 128);
        k_gemm_bf16<<<g, 256, 0, stream>>>(A, lda, Wd[wi], specs[wi].Kp, Bd[wi],
                                           Cf, Cb, ldc, M, N, Kp, relu);
    };
    auto ew = [&](size_t total) { return dim3((unsigned)((total + 255) / 256)); };

    // ---- stage 0: weights, zero-init
    for (int i = 0; i < ns; ++i) {
        int Np = (specs[i].N + 15) & ~15;
        size_t n = (size_t)Np * specs[i].Kp;
        k_pad_w<<<ew(n), 256, 0, stream>>>(Pf(specs[i].wi), Wd[i],
                                           specs[i].N, specs[i].K, Np, specs[i].Kp);
        k_pad_b<<<1, 64, 0, stream>>>(Pf(specs[i].bi), Bd[i], specs[i].N, Np);
    }
    zero(AMSG, (size_t)E * 352 * 2);
    zero(A1,   (size_t)E * 160 * 2);
    zero(A2,   (size_t)E * 160 * 2);
    zero(Vv,   (size_t)NN * 48 * 4);
    zero(CNT,  (size_t)NN * 4);
    k_cnt<<<ew(E), 256, 0, stream>>>(dst, CNT, E);

    // ---- edge embedding (geometry + we GVP)
    k_edge_geom<<<ew(E), 256, 0, stream>>>(pos, src, dst, Pf(WELNW), Pf(WELNB),
                                           Pf(WE + 0), AWE, AMSG, EVN, E);
    gemm(AWE, 32, 1, nullptr, AMSG + 128, 352, E, 32, 32, 0);       // es -> AMSG[:,128:160]
    gemm(AMSG + 128, 352, 2, GATE, nullptr, 16, E, 1, 32, 0);       // we gate
    k_we_vout<<<ew(E), 256, 0, stream>>>(EVN, GATE, Pf(WE + 5), Pf(WE + 0), EVOUT, E);

    // ---- node embedding: LN(atoms) @ wv_ws
    k_ln128_b16<<<ew((size_t)NN * 32), 256, 0, stream>>>(atoms, Pf(WVLNW), Pf(WVLNB), NB, NN);
    gemm(NB, 128, 0, S, nullptr, 128, NN, 128, 128, 0);

    // ---- message passing layers
    for (int l = 0; l < 5; ++l) {
        int Lb = 3 + l * 34;
        int g0 = 3 + l * 10;
        // gather edge inputs
        k_gather_s<<<ew((size_t)E * 128), 256, 0, stream>>>(S, src, dst, AMSG, 352, E);
        k_gather_v<<<ew((size_t)E * 33), 256, 0, stream>>>(Vv, EVOUT, src, dst, MV, E);
        // msg0 (vi=33,h=33,vo=16)
        k_combine<<<9, 64, 0, stream>>>(Pf(Lb + 21), Pf(Lb + 16), WC, 16, 33, 33);
        k_vn<<<ew((size_t)E * 33), 256, 0, stream>>>(MV, 33, Pf(Lb + 16), 33, AMSG, 352, 289, E);
        gemm(AMSG, 352, g0 + 0, nullptr, A1, 160, E, 128, 352, 0);
        gemm(A1, 160, g0 + 1, GATE, nullptr, 16, E, 16, 128, 0);
        k_relu_b16<<<ew((size_t)E * 128), 256, 0, stream>>>(A1, 160, E, 128);
        k_vout<<<ew((size_t)E * 16), 256, 0, stream>>>(MV, 33, WC, 16, GATE, 16, MVA, E);
        // msg1 (vi=16,h=16,vo=16)
        k_combine<<<4, 64, 0, stream>>>(Pf(Lb + 27), Pf(Lb + 22), WC, 16, 16, 16);
        k_vn<<<ew((size_t)E * 16), 256, 0, stream>>>(MVA, 16, Pf(Lb + 22), 16, A1, 160, 128, E);
        gemm(A1, 160, g0 + 2, nullptr, A2, 160, E, 128, 160, 0);
        gemm(A2, 160, g0 + 3, GATE, nullptr, 16, E, 16, 128, 0);
        k_relu_b16<<<ew((size_t)E * 128), 256, 0, stream>>>(A2, 160, E, 128);
        k_vout<<<ew((size_t)E * 16), 256, 0, stream>>>(MVA, 16, WC, 16, GATE, 16, MVB, E);
        // msg2 (no act)
        k_combine<<<4, 64, 0, stream>>>(Pf(Lb + 33), Pf(Lb + 28), WC, 16, 16, 16);
        k_vn<<<ew((size_t)E * 16), 256, 0, stream>>>(MVB, 16, Pf(Lb + 28), 16, A2, 160, 128, E);
        gemm(A2, 160, g0 + 4, MSOUT, nullptr, 128, E, 128, 160, 0);
        k_f2b<<<ew((size_t)E * 128), 256, 0, stream>>>(MSOUT, 128, A1, 160, E, 128);
        gemm(A1, 160, g0 + 5, GATE, nullptr, 16, E, 16, 128, 0);
        k_vout<<<ew((size_t)E * 16), 256, 0, stream>>>(MVB, 16, WC, 16, GATE, 16, MV, E); // MV reused [E,48]
        // scatter-mean + residual + ln0
        zero(AGGS, (size_t)NN * 128 * 4);
        zero(AGGV, (size_t)NN * 48 * 4);
        k_scatter_s<<<ew((size_t)E * 128), 256, 0, stream>>>(MSOUT, dst, AGGS, E);
        k_scatter_v<<<ew((size_t)E * 48), 256, 0, stream>>>(MV, dst, AGGV, E);
        k_resid_lnsv<<<ew((size_t)NN * 32), 256, 0, stream>>>(S, Vv, AGGS, AGGV, CNT,
                                                              Pf(Lb + 13), Pf(Lb + 12), NN);
        // ff0 (vi=16,h=32,vo=32), relu
        k_f2b<<<ew((size_t)NN * 128), 256, 0, stream>>>(S, 128, AFF0, 160, NN, 128);
        k_combine<<<8, 64, 0, stream>>>(Pf(Lb + 5), Pf(Lb + 0), WC, 32, 32, 16);
        k_vn<<<ew((size_t)NN * 32), 256, 0, stream>>>(Vv, 16, Pf(Lb + 0), 32, AFF0, 160, 128, NN);
        gemm(AFF0, 160, g0 + 6, nullptr, AFF1, 544, NN, 512, 160, 0);
        gemm(AFF1, 544, g0 + 7, GATEN, nullptr, 32, NN, 32, 512, 0);
        k_relu_b16<<<ew((size_t)NN * 512), 256, 0, stream>>>(AFF1, 544, NN, 512);
        k_vout<<<ew((size_t)NN * 32), 256, 0, stream>>>(Vv, 16, WC, 32, GATEN, 32, FV0, NN);
        // ff1 (vi=32,h=32,vo=16), no act
        k_combine<<<8, 64, 0, stream>>>(Pf(Lb + 11), Pf(Lb + 6), WC, 16, 32, 32);
        k_vn<<<ew((size_t)NN * 32), 256, 0, stream>>>(FV0, 32, Pf(Lb + 6), 32, AFF1, 544, 512, NN);
        gemm(AFF1, 544, g0 + 8, FS, nullptr, 128, NN, 128, 544, 0);
        k_f2b<<<ew((size_t)NN * 128), 256, 0, stream>>>(FS, 128, NB, 128, NN, 128);
        gemm(NB, 128, g0 + 9, GATEN, nullptr, 32, NN, 16, 128, 0);
        k_vout<<<ew((size_t)NN * 16), 256, 0, stream>>>(FV0, 32, WC, 16, GATEN, 32, FV, NN);
        // residual + ln1
        k_resid_lnsv<<<ew((size_t)NN * 32), 256, 0, stream>>>(S, Vv, FS, FV, nullptr,
                                                              Pf(Lb + 15), Pf(Lb + 14), NN);
    }

    // ---- prediction head
    k_f2b<<<ew((size_t)NN * 128), 256, 0, stream>>>(S, 128, NB, 128, NN, 128);
    gemm(NB, 128, 53, nullptr, NB2, 128, NN, 128, 128, 1);           // pred1 + relu
    gemm(NB2, 128, 54, LOGIT, nullptr, 16, NN, 2, 128, 0);           // pred2
    k_softmax2<<<ew(NN), 256, 0, stream>>>(LOGIT, 16, (float*)d_out, NN);
}
